// RingStarBlock_46179488367257
// MI455X (gfx1250) — matrix operand order repaired and checked by reading, not compiled, not run on hardware
//
#include <hip/hip_runtime.h>
#include <math.h>

#define B_    8
#define L_    2048
#define N_    512
#define H_    16
#define C_    16
#define D_    64
#define TOPK_ 45
#define BL_   (B_ * L_)

typedef unsigned short u16;
typedef unsigned int   u32;

typedef __attribute__((ext_vector_type(16))) __bf16 v16bf;
typedef __attribute__((ext_vector_type(8)))  float  v8f;
typedef __attribute__((ext_vector_type(2)))  float  v2f;
typedef __attribute__((__vector_size__(16))) int    i32x4;

#if defined(__has_builtin)
#if __has_builtin(__builtin_amdgcn_wmma_f32_16x16x4_f32)
#define HAVE_WMMA_F32X4 1
#endif
#if __has_builtin(__builtin_amdgcn_global_load_async_to_lds_b128) && \
    __has_builtin(__builtin_amdgcn_s_wait_asynccnt)
#define HAVE_ASYNC_LDS 1
#endif
#endif

union FragAB { v16bf v; u32 u[8]; };

__device__ inline u16 f2bf(float f) {
  union { float f; u32 u; } c; c.f = f;
  u32 r = c.u + 0x7FFFu + ((c.u >> 16) & 1u);
  return (u16)(r >> 16);
}
__device__ inline float bf2f(u16 h) {
  union { float f; u32 u; } c; c.u = ((u32)h) << 16;
  return c.f;
}

#ifdef HAVE_ASYNC_LDS
__device__ inline void async_cp16(const u16* g, u16* l) {
  __builtin_amdgcn_global_load_async_to_lds_b128(
      (__attribute__((address_space(1))) i32x4*)g,
      (__attribute__((address_space(3))) i32x4*)l, 0, 0);
}
#endif

// ---------------- small utility kernels ----------------

__global__ void f2bf_kernel(const float* __restrict__ src, u16* __restrict__ dst, int count) {
  int i = blockIdx.x * 256 + threadIdx.x;
  if (i < count) dst[i] = f2bf(src[i]);
}

// normed[l,n] = mean_b(x) / (std_b(x, ddof=1) + 1e-5)
__global__ void normed_kernel(const float* __restrict__ x, float* __restrict__ normed) {
  int i = blockIdx.x * 256 + threadIdx.x; // over L_*N_
  float s = 0.f, s2 = 0.f;
  for (int b = 0; b < B_; ++b) {
    float v = x[(size_t)b * (L_ * N_) + i];
    s += v; s2 += v * v;
  }
  float mean = s * (1.f / B_);
  float var = (s2 - (float)B_ * mean * mean) * (1.f / (B_ - 1));
  float stdv = sqrtf(fmaxf(var, 0.f)) + 1e-5f;
  normed[i] = mean / stdv;
}

// sim = normed^T @ normed  (N x N, K = L), fp32 WMMA 16x16x4 if available
__global__ void __launch_bounds__(128) simgemm_kernel(const float* __restrict__ normed,
                                                      float* __restrict__ sim) {
  __shared__ float As[64][18];
  __shared__ float Bs[64][18];
  int t = threadIdx.x;
  int wave = t >> 5, lane = t & 31, g = lane >> 4, ln = lane & 15;
  int wm = (wave >> 1) * 32, wn = (wave & 1) * 32;
  int tm = blockIdx.y * 64, tn = blockIdx.x * 64;
  v8f acc[2][2];
  for (int i = 0; i < 2; ++i) for (int j = 0; j < 2; ++j) for (int r = 0; r < 8; ++r) acc[i][j][r] = 0.f;

  int lrow = t >> 1, lbase = (t & 1) * 8;
  for (int kc = 0; kc < L_; kc += 16) {
    for (int i = 0; i < 8; ++i) {
      int k = lbase + i;
      As[lrow][k] = normed[(size_t)(kc + k) * N_ + tm + lrow];
      Bs[lrow][k] = normed[(size_t)(kc + k) * N_ + tn + lrow];
    }
    __syncthreads();
#ifdef HAVE_WMMA_F32X4
    for (int s = 0; s < 4; ++s) {
      for (int fm = 0; fm < 2; ++fm) {
        v2f a = *(const v2f*)&As[wm + fm * 16 + ln][s * 4 + g * 2];
        for (int fn = 0; fn < 2; ++fn) {
          v2f b = *(const v2f*)&Bs[wn + fn * 16 + ln][s * 4 + g * 2];
          acc[fm][fn] = __builtin_amdgcn_wmma_f32_16x16x4_f32(
              false, a, false, b, (short)0, acc[fm][fn], false, false);
        }
      }
    }
#else
    for (int fm = 0; fm < 2; ++fm)
      for (int fn = 0; fn < 2; ++fn) {
        int rr = wm + fm * 16 + g * 8;
        int cc = wn + fn * 16 + ln;
        for (int r = 0; r < 8; ++r) {
          float s = 0.f;
          for (int k = 0; k < 16; ++k) s += As[rr + r][k] * Bs[cc][k];
          acc[fm][fn][r] += s;
        }
      }
#endif
    __syncthreads();
  }
  for (int fm = 0; fm < 2; ++fm)
    for (int fn = 0; fn < 2; ++fn)
      for (int r = 0; r < 8; ++r) {
        int rr = tm + wm + fm * 16 + g * 8 + r;
        int cc = tn + wn + fn * 16 + ln;
        sim[(size_t)rr * N_ + cc] = acc[fm][fn][r];
      }
}

// top-45 per row -> boolean mask (uint8)
__global__ void topk_kernel(const float* __restrict__ sim, unsigned char* __restrict__ mask) {
  int n = blockIdx.x, t = threadIdx.x; // 64 threads
  __shared__ float vals[N_];
  __shared__ float tv[64];
  __shared__ int   ti[64];
  for (int m = t; m < N_; m += 64) {
    vals[m] = (m == n) ? -3.0e38f : sim[(size_t)n * N_ + m];
    mask[(size_t)n * N_ + m] = 0;
  }
  __syncthreads();
  for (int it = 0; it < TOPK_; ++it) {
    float lv = -3.0e38f; int li = 0;
    for (int j = 0; j < 8; ++j) {
      int m = t * 8 + j;
      float v = vals[m];
      if (v > lv) { lv = v; li = m; }
    }
    tv[t] = lv; ti[t] = li;
    __syncthreads();
    if (t == 0) {
      float bv = -3.0e38f; int bi = 0;
      for (int i = 0; i < 64; ++i) if (tv[i] > bv) { bv = tv[i]; bi = ti[i]; }
      vals[bi] = -3.0e38f;
      mask[(size_t)n * N_ + bi] = 1;
    }
    __syncthreads();
  }
}

// per (b,n): temporal stats, var_fused, Q, K, importance, summary
__global__ void router_kernel(const float* __restrict__ x, const float* __restrict__ var_embed,
                              const float* __restrict__ temp_w, const float* __restrict__ temp_b,
                              const float* __restrict__ q_w, const float* __restrict__ q_b,
                              const float* __restrict__ k_w, const float* __restrict__ k_b,
                              const float* __restrict__ imp_w, const float* __restrict__ imp_b,
                              float* __restrict__ Qb, float* __restrict__ Kb,
                              float* __restrict__ impb, float* __restrict__ summary) {
  int bid = blockIdx.x;
  int b = bid >> 9, n = bid & (N_ - 1);
  int t = threadIdx.x; // 64
  __shared__ float rs[64], rs2[64], vf[32];
  float s = 0.f, s2 = 0.f;
  const float* xp = x + (size_t)b * L_ * N_ + n;
  for (int i = t; i < L_; i += 64) {
    float v = xp[(size_t)i * N_];
    s += v; s2 += v * v;
  }
  rs[t] = s; rs2[t] = s2;
  __syncthreads();
  for (int o = 32; o > 0; o >>= 1) {
    if (t < o) { rs[t] += rs[t + o]; rs2[t] += rs2[t + o]; }
    __syncthreads();
  }
  float mean = rs[0] * (1.f / L_);
  float var = (rs2[0] - (float)L_ * mean * mean) * (1.f / (L_ - 1));
  float stdv = sqrtf(var + 1e-5f);
  if (t == 0) summary[b * N_ + n] = mean;
  if (t < H_) {
    vf[t] = var_embed[n * H_ + t];
    vf[H_ + t] = mean * temp_w[t] + stdv * temp_w[H_ + t] + temp_b[t];
  }
  __syncthreads();
  if (t < H_) {
    float q = q_b[t];
    for (int j = 0; j < 2 * H_; ++j) q += vf[j] * q_w[j * H_ + t];
    Qb[((size_t)b * N_ + n) * H_ + t] = q;
  } else if (t < 2 * H_) {
    int h = t - H_;
    float k = k_b[h];
    for (int j = 0; j < 2 * H_; ++j) k += vf[j] * k_w[j * H_ + h];
    Kb[((size_t)b * N_ + n) * H_ + h] = k;
  } else if (t == 2 * H_) {
    float z = imp_b[0];
    for (int j = 0; j < 2 * H_; ++j) z += vf[j] * imp_w[j];
    impb[b * N_ + n] = 1.f / (1.f + expf(-z));
  }
}

// masked attention softmax row -> write attn^T in bf16 (for GEMM B-operand)
__global__ void attn_kernel(const float* __restrict__ Qb, const float* __restrict__ Kb,
                            const unsigned char* __restrict__ mask, u16* __restrict__ attnT) {
  int bid = blockIdx.x;
  int b = bid >> 9, n = bid & (N_ - 1);
  int t = threadIdx.x; // 128
  __shared__ float qv[H_];
  __shared__ float red[128];
  if (t < H_) qv[t] = Qb[((size_t)b * N_ + n) * H_ + t];
  __syncthreads();
  float sc[4]; bool ok[4];
  float lmax = -3.0e38f;
  for (int j = 0; j < 4; ++j) {
    int m = t + j * 128;
    bool mk = mask[(size_t)n * N_ + m] != 0;
    float s = -3.0e38f;
    if (mk) {
      const float* kp = Kb + ((size_t)b * N_ + m) * H_;
      s = 0.f;
      for (int h = 0; h < H_; ++h) s += qv[h] * kp[h];
      s *= 0.25f; // 1/sqrt(H)
      if (s > lmax) lmax = s;
    }
    sc[j] = s; ok[j] = mk;
  }
  red[t] = lmax;
  __syncthreads();
  for (int o = 64; o > 0; o >>= 1) { if (t < o) red[t] = fmaxf(red[t], red[t + o]); __syncthreads(); }
  float gmax = red[0];
  __syncthreads();
  float ev[4]; float ls = 0.f;
  for (int j = 0; j < 4; ++j) { ev[j] = ok[j] ? expf(sc[j] - gmax) : 0.f; ls += ev[j]; }
  red[t] = ls;
  __syncthreads();
  for (int o = 64; o > 0; o >>= 1) { if (t < o) red[t] += red[t + o]; __syncthreads(); }
  float inv = 1.f / red[0];
  for (int j = 0; j < 4; ++j) {
    int m = t + j * 128;
    attnT[((size_t)b * N_ + m) * N_ + n] = f2bf(ev[j] * inv);
  }
}

__global__ void cwsoftmax_kernel(const float* __restrict__ cs, float* __restrict__ cw) {
  int n = blockIdx.x * blockDim.x + threadIdx.x;
  if (n >= N_) return;
  float mx = -3.0e38f;
  for (int c = 0; c < C_; ++c) mx = fmaxf(mx, cs[n * C_ + c]);
  float e[C_]; float s = 0.f;
  for (int c = 0; c < C_; ++c) { e[c] = expf(cs[n * C_ + c] - mx); s += e[c]; }
  float inv = 1.f / s;
  for (int c = 0; c < C_; ++c) cw[n * C_ + c] = e[c] * inv;
}

__global__ void centers_kernel(const float* __restrict__ summary, const float* __restrict__ cg_w,
                               const float* __restrict__ cg_b, float* __restrict__ ct) {
  int i = blockIdx.x * 256 + threadIdx.x; // B_*C_*D_ = 8192
  int b = i >> 10, cd = i & 1023;
  float s = cg_b[cd];
  for (int nn = 0; nn < N_; ++nn) s += summary[b * N_ + nn] * cg_w[(size_t)nn * (C_ * D_) + cd];
  ct[i] = s;
}

// A_center[b,c,d] = (ct[b,c,:] + varproj_b) @ cp1[:,d] + cp1_b[d]
__global__ void acenter_kernel(const float* __restrict__ ct, const float* __restrict__ varproj_b,
                               const float* __restrict__ cp1, const float* __restrict__ cp1_b,
                               float* __restrict__ ac) {
  int i = blockIdx.x * 256 + threadIdx.x; // 8192
  int bc = i >> 6, d = i & 63;
  float s = cp1_b[d];
  for (int j = 0; j < D_; ++j) s += (ct[bc * D_ + j] + varproj_b[j]) * cp1[j * D_ + d];
  ac[i] = s;
}

// collapse cp2 row-sums (einsum sums over d)
__global__ void w2s_kernel(const float* __restrict__ cp2, const float* __restrict__ cp2_b,
                           float* __restrict__ w2s, float* __restrict__ b2s) {
  int d = threadIdx.x; // 64
  float s = 0.f;
  for (int j = 0; j < D_; ++j) s += cp2[d * D_ + j];
  w2s[d] = s;
  if (d == 0) {
    float t = 0.f;
    for (int j = 0; j < D_; ++j) t += cp2_b[j];
    *b2s = t;
  }
}

// Wbig[n,d] = (cw @ varproj @ cp1)[n,d]  in bf16
__global__ void wbig_kernel(const float* __restrict__ cw, const float* __restrict__ varproj,
                            const float* __restrict__ cp1, u16* __restrict__ wbig) {
  int n = blockIdx.x, d = threadIdx.x; // 64
  __shared__ float t1[D_];
  float s = 0.f;
  for (int c = 0; c < C_; ++c) s += cw[n * C_ + c] * varproj[c * D_ + d];
  t1[d] = s;
  __syncthreads();
  float w = 0.f;
  for (int j = 0; j < D_; ++j) w += t1[j] * cp1[j * D_ + d];
  wbig[n * D_ + d] = f2bf(w);
}

// center path: s[b,l,c] = sum_d gelu(A_center + xc1)*w2s + b2s ; ctv = s @ cw^T -> cat[:,512:1024]
__global__ void ctv_kernel(const float* __restrict__ xc1, const float* __restrict__ ac,
                           const float* __restrict__ w2s, const float* __restrict__ b2s,
                           const float* __restrict__ cw, u16* __restrict__ cat) {
  int idx = blockIdx.x;           // over BL_
  int b = idx >> 11;              // L_ = 2048
  int t = threadIdx.x;            // 64
  __shared__ float red2[C_ * D_];
  __shared__ float sc[C_];
  float xd = xc1[(size_t)idx * D_ + t];
  float wd = w2s[t];
  for (int c = 0; c < C_; ++c) {
    float v = ac[(size_t)(b * C_ + c) * D_ + t] + xd;
    float g = 0.5f * v * (1.f + erff(v * 0.70710678118654752f));
    red2[c * D_ + t] = g * wd;
  }
  __syncthreads();
  if (t < C_) {
    float s = b2s[0];
    for (int d2 = 0; d2 < D_; ++d2) s += red2[t * D_ + d2];
    sc[t] = s;
  }
  __syncthreads();
  for (int r = 0; r < 8; ++r) {
    int n = t + r * 64;
    float o = 0.f;
    for (int c = 0; c < C_; ++c) o += sc[c] * cw[n * C_ + c];
    cat[(size_t)idx * 1024 + 512 + n] = f2bf(o);
  }
}

// final LayerNorm (in-place on d_out; h already written there)
__global__ void ln_kernel(float* __restrict__ out, const float* __restrict__ ln_g,
                          const float* __restrict__ ln_b) {
  int row = blockIdx.x, t = threadIdx.x; // 128
  __shared__ float r1[128], r2[128];
  float* p = out + (size_t)row * N_;
  float v[4]; float s = 0.f, s2 = 0.f;
  for (int j = 0; j < 4; ++j) { v[j] = p[t + j * 128]; s += v[j]; s2 += v[j] * v[j]; }
  r1[t] = s; r2[t] = s2;
  __syncthreads();
  for (int o = 64; o > 0; o >>= 1) { if (t < o) { r1[t] += r1[t + o]; r2[t] += r2[t + o]; } __syncthreads(); }
  float mu = r1[0] * (1.f / N_);
  float var = r2[0] * (1.f / N_) - mu * mu;
  float inv = rsqrtf(var + 1e-5f);
  for (int j = 0; j < 4; ++j) {
    int n = t + j * 128;
    p[n] = (v[j] - mu) * inv * ln_g[n] + ln_b[n];
  }
}

// ---------------- main bf16 WMMA GEMM (64x64 tile, 4 waves, K-chunk 32) ----------------
// Double-buffered LDS staging; async global->LDS copies (ASYNCcnt) when available.
// A staged row-major [64][40] (16B-aligned rows for b128 async).
// B staged un-transposed [32][72]; transpose happens in the fragment read
// (K-major per ISA layout: element pair 2v,2v+1 at K = (v>>2)*16 + g*8 + 2*(v&3)).
// EPI 0: out_f[row*N+col] = acc                              (xc1)
// EPI 1: ring epilogue -> cat[:, :512] bf16                  (routing, batched over z)
// EPI 2: gate epilogue -> fused bf16                         (gate)
// EPI 3: +fus_b +x -> out_f (pre-LN h)                       (fuse)
template <int EPI>
__global__ void __launch_bounds__(128) gemm64_kernel(
    const u16* __restrict__ Ag, const u16* __restrict__ Bg,
    int M, int N, int K, unsigned long long batchA, unsigned long long batchB,
    const float* __restrict__ aux0, const float* __restrict__ aux1,
    const u16* __restrict__ catb, u16* __restrict__ outb, float* __restrict__ outf) {
  __shared__ u16 As[2][64][40];
  __shared__ u16 Bs[2][32][72];
  int t = threadIdx.x;
  int wave = t >> 5, lane = t & 31, g = lane >> 4, ln = lane & 15;
  int wm = (wave >> 1) * 32, wn = (wave & 1) * 32;
  int tm = blockIdx.y * 64, tn = blockIdx.x * 64;
  int bz = blockIdx.z;
  const u16* Ab = Ag + (unsigned long long)bz * batchA;
  const u16* Bb = Bg + (unsigned long long)bz * batchB;

  v8f acc[2][2];
  for (int i = 0; i < 2; ++i) for (int j = 0; j < 2; ++j) for (int r = 0; r < 8; ++r) acc[i][j][r] = 0.f;

  int arow = t >> 1, aoff = (t & 1) * 16;   // A: 64 rows x 32 u16, 32B per thread
  int brow = t >> 2, boff = (t & 3) * 16;   // B: 32 rows x 64 u16, 32B per thread

  auto stage = [&](int bb, int kc2) {
    const u16* ga = Ab + (size_t)(tm + arow) * K + kc2 + aoff;
    u16* la = &As[bb][arow][aoff];
    const u16* gb = Bb + (size_t)(kc2 + brow) * N + tn + boff;
    u16* lb = &Bs[bb][brow][boff];
#ifdef HAVE_ASYNC_LDS
    async_cp16(ga, la);
    async_cp16(ga + 8, la + 8);
    async_cp16(gb, lb);
    async_cp16(gb + 8, lb + 8);
#else
    for (int j = 0; j < 8; ++j) ((u32*)la)[j] = ((const u32*)ga)[j];
    for (int j = 0; j < 8; ++j) ((u32*)lb)[j] = ((const u32*)gb)[j];
    __builtin_prefetch(ga + 64, 0, 1);
#endif
  };

  int buf = 0;
  stage(0, 0);
  for (int kc = 0; kc < K; kc += 32) {
#ifdef HAVE_ASYNC_LDS
    __builtin_amdgcn_s_wait_asynccnt(0);
#endif
    __syncthreads();
    if (kc + 32 < K) stage(buf ^ 1, kc + 32);

    FragAB a[2], b[2];
    for (int fm = 0; fm < 2; ++fm) {
      int row = wm + fm * 16 + ln;
      for (int v = 0; v < 8; ++v) {
        int kk2 = ((v >> 2) << 4) + (g << 3) + ((v & 3) << 1);
        a[fm].u[v] = *(const u32*)&As[buf][row][kk2];
      }
    }
    for (int fn = 0; fn < 2; ++fn) {
      int col = wn + fn * 16 + ln;
      for (int v = 0; v < 8; ++v) {
        int k0 = ((v >> 2) << 4) + (g << 3) + ((v & 3) << 1);
        u32 lo = Bs[buf][k0][col];
        u32 hi = Bs[buf][k0 + 1][col];
        b[fn].u[v] = lo | (hi << 16);
      }
    }
    for (int fm = 0; fm < 2; ++fm)
      for (int fn = 0; fn < 2; ++fn)
        acc[fm][fn] = __builtin_amdgcn_wmma_f32_16x16x32_bf16(
            false, a[fm].v, false, b[fn].v, (short)0, acc[fm][fn], false, false);
    buf ^= 1;
  }

  for (int fm = 0; fm < 2; ++fm)
    for (int fn = 0; fn < 2; ++fn)
      for (int r = 0; r < 8; ++r) {
        int lrow = tm + wm + fm * 16 + g * 8 + r;
        int col = tn + wn + fn * 16 + ln;
        float a0 = acc[fm][fn][r];
        if (EPI == 0) {
          outf[(size_t)lrow * N + col] = a0;
        } else if (EPI == 1) {
          size_t grow = (size_t)bz * M + lrow;
          float iv = aux0[bz * N_ + col];
          float xv = aux1[grow * N_ + col];
          float rv = iv * a0 + (1.f - iv) * xv;
          outb[grow * 1024 + col] = f2bf(rv);
        } else if (EPI == 2) {
          float gv = 1.f / (1.f + expf(-(a0 + aux0[col])));
          float ring = bf2f(catb[(size_t)lrow * 1024 + col]);
          float cv = bf2f(catb[(size_t)lrow * 1024 + 512 + col]);
          outb[(size_t)lrow * N_ + col] = f2bf(gv * ring + (1.f - gv) * cv);
        } else {
          float h = a0 + aux0[col] + aux1[(size_t)lrow * N_ + col];
          outf[(size_t)lrow * N_ + col] = h;
        }
      }
}

// ---------------- host launcher ----------------

extern "C" void kernel_launch(void* const* d_in, const int* in_sizes, int n_in,
                              void* d_out, int out_size, void* d_ws, size_t ws_size,
                              hipStream_t stream) {
  (void)in_sizes; (void)n_in; (void)out_size; (void)ws_size;
  const float* x            = (const float*)d_in[0];
  const float* var_embed    = (const float*)d_in[1];
  const float* temp_w       = (const float*)d_in[2];
  const float* temp_b       = (const float*)d_in[3];
  const float* q_w          = (const float*)d_in[4];
  const float* q_b          = (const float*)d_in[5];
  const float* k_w          = (const float*)d_in[6];
  const float* k_b          = (const float*)d_in[7];
  const float* imp_w        = (const float*)d_in[8];
  const float* imp_b        = (const float*)d_in[9];
  const float* cluster_score= (const float*)d_in[10];
  const float* varproj_w    = (const float*)d_in[11];
  const float* varproj_b    = (const float*)d_in[12];
  const float* cp1_w        = (const float*)d_in[13];
  const float* cp1_b        = (const float*)d_in[14];
  const float* cp2_w        = (const float*)d_in[15];
  const float* cp2_b        = (const float*)d_in[16];
  const float* cg_w         = (const float*)d_in[17];
  const float* cg_b         = (const float*)d_in[18];
  const float* gate_w       = (const float*)d_in[19];
  const float* gate_b       = (const float*)d_in[20];
  const float* fus_w        = (const float*)d_in[21];
  const float* fus_b        = (const float*)d_in[22];
  const float* ln_g         = (const float*)d_in[23];
  const float* ln_b         = (const float*)d_in[24];
  float* out = (float*)d_out;

  char* w = (char*)d_ws;
  size_t off = 0;
  auto give = [&](size_t bytes) -> char* {
    char* p = w + off;
    off += (bytes + 255) & ~(size_t)255;
    return p;
  };
  float* normed  = (float*)give((size_t)L_ * N_ * 4);
  float* sim     = (float*)give((size_t)N_ * N_ * 4);
  unsigned char* mask = (unsigned char*)give((size_t)N_ * N_);
  float* Qb      = (float*)give((size_t)B_ * N_ * H_ * 4);
  float* Kb      = (float*)give((size_t)B_ * N_ * H_ * 4);
  float* impb    = (float*)give((size_t)B_ * N_ * 4);
  float* summary = (float*)give((size_t)B_ * N_ * 4);
  u16* x_bf      = (u16*)give((size_t)BL_ * N_ * 2);
  u16* attnT     = (u16*)give((size_t)B_ * N_ * N_ * 2);
  u16* cat       = (u16*)give((size_t)BL_ * 1024 * 2);
  u16* gw_bf     = (u16*)give((size_t)2 * N_ * N_ * 2);
  u16* fw_bf     = (u16*)give((size_t)N_ * N_ * 2);
  u16* fused_bf  = (u16*)give((size_t)BL_ * N_ * 2);
  float* cw      = (float*)give((size_t)N_ * C_ * 4);
  float* ct      = (float*)give((size_t)B_ * C_ * D_ * 4);
  float* ac      = (float*)give((size_t)B_ * C_ * D_ * 4);
  float* w2s     = (float*)give((size_t)D_ * 4);
  float* b2s     = (float*)give(4);
  u16* wbig      = (u16*)give((size_t)N_ * D_ * 2);
  float* xc1     = (float*)give((size_t)BL_ * D_ * 4);

  // conversions
  f2bf_kernel<<<(BL_ * N_ + 255) / 256, 256, 0, stream>>>(x, x_bf, BL_ * N_);
  f2bf_kernel<<<(2 * N_ * N_ + 255) / 256, 256, 0, stream>>>(gate_w, gw_bf, 2 * N_ * N_);
  f2bf_kernel<<<(N_ * N_ + 255) / 256, 256, 0, stream>>>(fus_w, fw_bf, N_ * N_);

  // static mask
  normed_kernel<<<(L_ * N_) / 256, 256, 0, stream>>>(x, normed);
  simgemm_kernel<<<dim3(N_ / 64, N_ / 64), 128, 0, stream>>>(normed, sim);
  topk_kernel<<<N_, 64, 0, stream>>>(sim, mask);

  // router
  router_kernel<<<B_ * N_, 64, 0, stream>>>(x, var_embed, temp_w, temp_b, q_w, q_b, k_w, k_b,
                                            imp_w, imp_b, Qb, Kb, impb, summary);
  attn_kernel<<<B_ * N_, 128, 0, stream>>>(Qb, Kb, mask, attnT);

  // center-path precomputation
  cwsoftmax_kernel<<<2, 256, 0, stream>>>(cluster_score, cw);
  centers_kernel<<<(B_ * C_ * D_) / 256, 256, 0, stream>>>(summary, cg_w, cg_b, ct);
  acenter_kernel<<<(B_ * C_ * D_) / 256, 256, 0, stream>>>(ct, varproj_b, cp1_w, cp1_b, ac);
  w2s_kernel<<<1, 64, 0, stream>>>(cp2_w, cp2_b, w2s, b2s);
  wbig_kernel<<<N_, 64, 0, stream>>>(cw, varproj_w, cp1_w, wbig);

  // xc1 = x @ Wbig   (M=BL, N=64, K=512)
  gemm64_kernel<0><<<dim3(1, BL_ / 64, 1), 128, 0, stream>>>(
      x_bf, wbig, BL_, D_, N_, 0ull, 0ull, nullptr, nullptr, nullptr, nullptr, xc1);

  // ring: per batch x[b] @ attnT[b] with importance-blend epilogue -> cat[:, :512]
  gemm64_kernel<1><<<dim3(N_ / 64, L_ / 64, B_), 128, 0, stream>>>(
      x_bf, attnT, L_, N_, N_, (unsigned long long)L_ * N_, (unsigned long long)N_ * N_,
      impb, x, nullptr, cat, nullptr);

  // center_to_var -> cat[:, 512:1024]
  ctv_kernel<<<BL_, 64, 0, stream>>>(xc1, ac, w2s, b2s, cw, cat);

  // gate GEMM (K=1024) with sigmoid-blend epilogue -> fused bf16
  gemm64_kernel<2><<<dim3(N_ / 64, BL_ / 64, 1), 128, 0, stream>>>(
      cat, gw_bf, BL_, N_, 2 * N_, 0ull, 0ull, gate_b, nullptr, cat, fused_bf, nullptr);

  // fuse GEMM + bias + residual -> h in d_out
  gemm64_kernel<3><<<dim3(N_ / 64, BL_ / 64, 1), 128, 0, stream>>>(
      fused_bf, fw_bf, BL_, N_, N_, 0ull, 0ull, fus_b, x, nullptr, nullptr, out);

  // layernorm in place
  ln_kernel<<<BL_, 128, 0, stream>>>(out, ln_g, ln_b);
}
